// FlowNetC_22600117911679
// MI455X (gfx1250) — matrix operand order repaired and checked by reading, not compiled, not run on hardware
//
#include <hip/hip_runtime.h>
#include <hip/hip_bf16.h>

typedef __attribute__((ext_vector_type(16))) __bf16 v16bf;
typedef __attribute__((ext_vector_type(8)))  float  v8f;
typedef __attribute__((ext_vector_type(4)))  int    v4i;

union Frag { v16bf v; unsigned int u[8]; };

#if defined(__HIP_DEVICE_COMPILE__) && \
    __has_builtin(__builtin_amdgcn_global_load_async_to_lds_b128) && \
    __has_builtin(__builtin_amdgcn_s_wait_asynccnt)
#define USE_ASYNC_LDS 1
// builtin signature (from hipcc diagnostic): (int4 AS1* gaddr, int4 AS3* lds, imm, imm)
#define AS1(p) ((__attribute__((address_space(1))) v4i*)(void*)(p))
#define AS3(p) ((__attribute__((address_space(3))) v4i*)(void*)(p))
#else
#define USE_ASYNC_LDS 0
#endif

__device__ __forceinline__ unsigned short f2bf(float f) {
    union { float f; unsigned int u; } x; x.f = f;
    unsigned int u = x.u;
    u += 0x7fffu + ((u >> 16) & 1u);        // round-to-nearest-even
    return (unsigned short)(u >> 16);
}
__device__ __forceinline__ float bf2f(unsigned short h) {
    union { float f; unsigned int u; } x; x.u = ((unsigned int)h) << 16; return x.f;
}

// ---------------------------------------------------------------------------
// Zero fill (dwords)
// ---------------------------------------------------------------------------
__global__ void zero_u32(unsigned int* __restrict__ p, long n) {
    long i = (long)blockIdx.x * blockDim.x + threadIdx.x;
    if (i < n) p[i] = 0u;
}

// ---------------------------------------------------------------------------
// Weight conversion: OIHW fp32 -> [O][kh*kw][Kc] bf16 (Kc >= Cin, zero pad)
// ---------------------------------------------------------------------------
__global__ void cvt_weight(const float* __restrict__ src, unsigned short* __restrict__ dst,
                           int Cout, int Cin, int KHW, int Kc) {
    long idx = (long)blockIdx.x * blockDim.x + threadIdx.x;
    long total = (long)Cout * KHW * Kc;
    if (idx >= total) return;
    int i = (int)(idx % Kc);
    long t = idx / Kc;
    int r = (int)(t % KHW);
    int o = (int)(t / KHW);
    float v = (i < Cin) ? src[((long)o * Cin + i) * KHW + r] : 0.0f;
    dst[idx] = f2bf(v);
}

// ---------------------------------------------------------------------------
// conv1: 3->64, k7 s2 p3, fp32 NCHW in -> bf16 NHWC out, ReLU. Cin=3 so a
// direct VALU conv; ~6% of FLOPs and image-read bandwidth bound anyway.
// ---------------------------------------------------------------------------
__global__ void conv1_direct(const float* __restrict__ X, const float* __restrict__ W,
                             const float* __restrict__ bias, unsigned short* __restrict__ Y) {
    long idx = (long)blockIdx.x * blockDim.x + threadIdx.x;
    const long NP = 8L * 192 * 256;
    if (idx >= NP * 64) return;
    int co = (int)(idx & 63);
    long p = idx >> 6;
    int wo = (int)(p % 256); long t = p / 256;
    int ho = (int)(t % 192); int b = (int)(t / 192);
    int hbase = ho * 2 - 3, wbase = wo * 2 - 3;
    float s = bias[co];
    for (int c = 0; c < 3; ++c) {
        const float* xc = X + ((long)(b * 3 + c)) * 384 * 512;
        const float* wc = W + (co * 3 + c) * 49;
        for (int ky = 0; ky < 7; ++ky) {
            int hin = hbase + ky;
            if (hin < 0 || hin >= 384) continue;
            const float* xr = xc + (long)hin * 512;
            for (int kx = 0; kx < 7; ++kx) {
                int win = wbase + kx;
                if (win < 0 || win >= 512) continue;
                s += xr[win] * wc[ky * 7 + kx];
            }
        }
    }
    s = s > 0.0f ? s : 0.0f;
    Y[p * 64 + co] = f2bf(s);
}

// ---------------------------------------------------------------------------
// Block-tiled implicit-GEMM conv via v_wmma_f32_16x16x32_bf16.
//   Block (256 thr = 8 waves) computes a 64(M=Cout) x 128(N=pos) tile.
//   Wave w: wM = w&1 picks M half (32 rows = 2 frags), wN = w>>1 picks a
//   32-wide N column (2 frags) -> 2x2 fragment grid, 4 WMMAs per K-chunk.
//   Per (ky,kx,K-chunk of 32): stage 64x32 A (weights) and 128x32 B (im2col)
//   bf16 tiles into LDS — via GLOBAL_LOAD_ASYNC_TO_LDS_B128 (ASYNCcnt) when
//   available, else coalesced b128 loads+stores. LDS rows padded to 40 halves.
//   Requires: Npos % 128 == 0 (true for all layers), Kc % 32 == 0.
// ---------------------------------------------------------------------------
__global__ __launch_bounds__(256) void conv_wmma(
    const unsigned short* __restrict__ X, const unsigned short* __restrict__ Wt,
    const float* __restrict__ bias,
    unsigned short* __restrict__ Y, float* __restrict__ Ynchw,
    int Hin, int Win, int Hout, int Wout, int Npos,
    int stride, int pad, int kh, int kw, int Cout, int Kc,
    int out_cs, int out_choff) {
    __shared__ unsigned short lA[64 * 40];    //  64 rows x 32 K (pad to 40)
    __shared__ unsigned short lB[128 * 40];   // 128 rows x 32 K (pad to 40)

    const int tid  = threadIdx.x;
    const int lane = tid & 31;
    const int wave = tid >> 5;
    const int wM = wave & 1;                 // M half within block tile
    const int wN = wave >> 1;                // N sub-tile (0..3), 32 wide
    const int ncol = lane & 15;
    const int hi   = lane >> 4;

    const int tilesM = (Cout + 63) >> 6;
    const int tM = blockIdx.x % tilesM;
    const int tN = blockIdx.x / tilesM;
    const int HW = Hout * Wout;

    // ---- cooperative-load coordinates (per thread) ----
    const int rr  = tid >> 2;                // A row 0..63 / B row pair base
    const int seg = tid & 3;                 // 16B segment 0..3
    int coA = tM * 64 + rr;                  // weight row
    if (coA >= Cout) coA = Cout - 1;         // clamp (unused rows, avoid OOB)
    // two B rows per thread: rr and rr+64
    const int pB0 = tN * 128 + rr;
    const int pB1 = pB0 + 64;
    const int bB0 = pB0 / HW; const int remB0 = pB0 - bB0 * HW;
    const int hoB0 = remB0 / Wout, woB0 = remB0 - hoB0 * Wout;
    const int hbB0 = hoB0 * stride - pad, wbB0 = woB0 * stride - pad;
    const int bB1 = pB1 / HW; const int remB1 = pB1 - bB1 * HW;
    const int hoB1 = remB1 / Wout, woB1 = remB1 - hoB1 * Wout;
    const int hbB1 = hoB1 * stride - pad, wbB1 = woB1 * stride - pad;

    v8f acc[2][2] = {};                      // [M-frag s][N-frag t]
    int  mbase[2]; bool mok[2];
    mbase[0] = tM * 64 + wM * 32; mbase[1] = mbase[0] + 16;
    mok[0] = mbase[0] < Cout;     mok[1] = mbase[1] < Cout;

    for (int ky = 0; ky < kh; ++ky) {
        const int hin0 = hbB0 + ky, hin1 = hbB1 + ky;
        for (int kx = 0; kx < kw; ++kx) {
            const int win0 = wbB0 + kx, win1 = wbB1 + kx;
            const bool ok0 = (hin0 >= 0) && (hin0 < Hin) && (win0 >= 0) && (win0 < Win);
            const bool ok1 = (hin1 >= 0) && (hin1 < Hin) && (win1 >= 0) && (win1 < Win);
            const unsigned short* xrow0 =
                X + ((long)((bB0 * Hin + hin0) * Win + win0)) * Kc + seg * 8;
            const unsigned short* xrow1 =
                X + ((long)((bB1 * Hin + hin1) * Win + win1)) * Kc + seg * 8;
            const unsigned short* wrow =
                Wt + ((long)((coA * kh + ky) * kw + kx)) * Kc + seg * 8;
            for (int cc = 0; cc < Kc; cc += 32) {
                __syncthreads();
#if USE_ASYNC_LDS
                // async global -> LDS staging (no VGPR round-trip); ASYNCcnt
                __builtin_amdgcn_global_load_async_to_lds_b128(
                    AS1(wrow + cc), AS3(&lA[rr * 40 + seg * 8]), 0, 0);
                if (ok0)
                    __builtin_amdgcn_global_load_async_to_lds_b128(
                        AS1(xrow0 + cc), AS3(&lB[rr * 40 + seg * 8]), 0, 0);
                else
                    *(uint4*)(&lB[rr * 40 + seg * 8]) = make_uint4(0u, 0u, 0u, 0u);
                if (ok1)
                    __builtin_amdgcn_global_load_async_to_lds_b128(
                        AS1(xrow1 + cc), AS3(&lB[(rr + 64) * 40 + seg * 8]), 0, 0);
                else
                    *(uint4*)(&lB[(rr + 64) * 40 + seg * 8]) = make_uint4(0u, 0u, 0u, 0u);
                __builtin_amdgcn_s_wait_asynccnt(0);
#else
                uint4 a4 = *(const uint4*)(wrow + cc);
                uint4 b40 = make_uint4(0u, 0u, 0u, 0u);
                uint4 b41 = make_uint4(0u, 0u, 0u, 0u);
                if (ok0) b40 = *(const uint4*)(xrow0 + cc);
                if (ok1) b41 = *(const uint4*)(xrow1 + cc);
                *(uint4*)(&lA[rr * 40 + seg * 8]) = a4;
                *(uint4*)(&lB[rr * 40 + seg * 8]) = b40;
                *(uint4*)(&lB[(rr + 64) * 40 + seg * 8]) = b41;
#endif
                if (cc + 32 < Kc) {
                    if (ok0) __builtin_prefetch(xrow0 + cc + 32, 0, 1);
                    if (ok1) __builtin_prefetch(xrow1 + cc + 32, 0, 1);
                }
                __syncthreads();

                // two B fragments (32x16): lane = column ncol, half picks K range
                Frag bfr[2];
#pragma unroll
                for (int t = 0; t < 2; ++t)
#pragma unroll
                    for (int j = 0; j < 8; ++j)
                        bfr[t].u[j] = *(const unsigned int*)(
                            &lB[(wN * 32 + t * 16 + ncol) * 40 + 16 * hi + 2 * j]);

                // two A fragments (16x32), 2x2 WMMA grid
#pragma unroll
                for (int s = 0; s < 2; ++s) {
                    if (!mok[s]) continue;            // wave-uniform guard
                    Frag af;
#pragma unroll
                    for (int j = 0; j < 8; ++j) {
                        int ka = 16 * hi + 2 * j;
                        af.u[j] = *(const unsigned int*)(
                            &lA[(wM * 32 + s * 16 + ncol) * 40 + ka]);
                    }
#pragma unroll
                    for (int t = 0; t < 2; ++t)
                        acc[s][t] = __builtin_amdgcn_wmma_f32_16x16x32_bf16(
                            false, af.v, false, bfr[t].v, (short)0, acc[s][t], false, false);
                }
            }
        }
    }

    // ---- epilogue: bias + ReLU; lane owns 8 consecutive channels per frag ----
#pragma unroll
    for (int t = 0; t < 2; ++t) {
        const int pT = tN * 128 + wN * 32 + t * 16 + ncol;
        const int bT = pT / HW; const int remT = pT - bT * HW;
        const int hoT = remT / Wout, woT = remT - hoT * Wout;
#pragma unroll
        for (int s = 0; s < 2; ++s) {
            if (!mok[s]) continue;
            unsigned int pk[4];
#pragma unroll
            for (int q = 0; q < 4; ++q) {
                int co0 = mbase[s] + 2 * q + 8 * hi;
                float v0 = acc[s][t][2 * q]     + bias[co0];
                float v1 = acc[s][t][2 * q + 1] + bias[co0 + 1];
                v0 = v0 > 0.0f ? v0 : 0.0f;
                v1 = v1 > 0.0f ? v1 : 0.0f;
                pk[q] = (unsigned int)f2bf(v0) | ((unsigned int)f2bf(v1) << 16);
                if (Ynchw) {
                    Ynchw[(((long)bT * Cout + co0)     * Hout + hoT) * Wout + woT] = v0;
                    Ynchw[(((long)bT * Cout + co0 + 1) * Hout + hoT) * Wout + woT] = v1;
                }
            }
            uint4 st; st.x = pk[0]; st.y = pk[1]; st.z = pk[2]; st.w = pk[3];
            *(uint4*)(Y + (long)pT * out_cs + out_choff + mbase[s] + 8 * hi) = st;
        }
    }
}

// ---------------------------------------------------------------------------
// Correlation: 441 shifts (21x21, stride 2, D=20) of f2 vs f1, mean over 256
// channels, ReLU, bf16 into concat buffer channels [32..473). One block per
// position; f1 row staged through LDS; per-thread shift dot products.
// ---------------------------------------------------------------------------
__global__ __launch_bounds__(448) void correlation_k(
    const unsigned short* __restrict__ F1, const unsigned short* __restrict__ F2,
    unsigned short* __restrict__ CAT) {
    __shared__ unsigned int s1[128];            // 256 bf16 = one f1 pixel row
    int p  = blockIdx.x;                         // [0, 8*48*64)
    int wo = p & 63; int t = p >> 6;
    int ho = t % 48; int b = t / 48;
    int tid = threadIdx.x;
    if (tid < 128) s1[tid] = ((const unsigned int*)F1)[(long)p * 128 + tid];
    __syncthreads();
    if (tid >= 441) return;
    int oy = (tid / 21) * 2, ox = (tid % 21) * 2;
    int h2 = ho + oy - 20, w2 = wo + ox - 20;
    float sum = 0.0f;
    if (h2 >= 0 && h2 < 48 && w2 >= 0 && w2 < 64) {
        const unsigned int* f2r =
            (const unsigned int*)(F2 + ((long)((b * 48 + h2) * 64 + w2)) * 256);
        for (int j = 0; j < 128; ++j) {
            unsigned int u1 = s1[j], u2 = f2r[j];
            sum += bf2f((unsigned short)(u1 & 0xffffu)) * bf2f((unsigned short)(u2 & 0xffffu));
            sum += bf2f((unsigned short)(u1 >> 16))     * bf2f((unsigned short)(u2 >> 16));
        }
    }
    sum *= (1.0f / 256.0f);
    sum = sum > 0.0f ? sum : 0.0f;
    CAT[(long)p * 480 + 32 + tid] = f2bf(sum);
}

// ---------------------------------------------------------------------------
// Host-side orchestration
// ---------------------------------------------------------------------------
extern "C" void kernel_launch(void* const* d_in, const int* in_sizes, int n_in,
                              void* d_out, int out_size, void* d_ws, size_t ws_size,
                              hipStream_t stream) {
    (void)in_sizes; (void)n_in; (void)out_size; (void)ws_size;
    const float* image1 = (const float*)d_in[0];
    const float* image2 = (const float*)d_in[1];
    // params dict order: w1,b1,w2,b2,w3,b3,w31,b31,w4,b4,w41,b41,w5,b5,w51,b51,w6,b6,w61,b61,wr,br
    const float* w1  = (const float*)d_in[2];  const float* b1  = (const float*)d_in[3];
    const float* w2  = (const float*)d_in[4];  const float* b2  = (const float*)d_in[5];
    const float* w3  = (const float*)d_in[6];  const float* b3  = (const float*)d_in[7];
    const float* w31 = (const float*)d_in[8];  const float* b31 = (const float*)d_in[9];
    const float* w4  = (const float*)d_in[10]; const float* b4  = (const float*)d_in[11];
    const float* w41 = (const float*)d_in[12]; const float* b41 = (const float*)d_in[13];
    const float* w5  = (const float*)d_in[14]; const float* b5  = (const float*)d_in[15];
    const float* w51 = (const float*)d_in[16]; const float* b51 = (const float*)d_in[17];
    const float* w6  = (const float*)d_in[18]; const float* b6  = (const float*)d_in[19];
    const float* w61 = (const float*)d_in[20]; const float* b61 = (const float*)d_in[21];
    const float* wr  = (const float*)d_in[22]; const float* br  = (const float*)d_in[23];

    float* out = (float*)d_out;
    float* out_c2a = out + 0;
    float* out_x31 = out + 12582912;   // 8*128*96*128
    float* out_x41 = out + 18874368;   // + 8*256*48*64
    float* out_x51 = out + 22020096;   // + 8*512*24*32
    float* out_x61 = out + 22806528;   // + 8*512*12*16

    char* ws = (char*)d_ws;
    size_t off = 0;
    auto alloc_bf = [&](long elems) -> unsigned short* {
        off = (off + 255) & ~(size_t)255;
        unsigned short* p = (unsigned short*)(ws + off);
        off += (size_t)elems * 2;
        return p;
    };
    // converted weights (bf16, K-major [O][khw][Kc])
    unsigned short* wb2  = alloc_bf(128L * 25 * 64);
    unsigned short* wb3  = alloc_bf(256L * 25 * 128);
    unsigned short* wbr  = alloc_bf(32L * 1 * 256);
    unsigned short* wb31 = alloc_bf(256L * 9 * 480);
    unsigned short* wb4  = alloc_bf(512L * 9 * 256);
    unsigned short* wb41 = alloc_bf(512L * 9 * 512);
    unsigned short* wb5  = alloc_bf(512L * 9 * 512);
    unsigned short* wb51 = alloc_bf(512L * 9 * 512);
    unsigned short* wb6  = alloc_bf(1024L * 9 * 512);
    unsigned short* wb61 = alloc_bf(1024L * 9 * 1024);
    // activations (bf16 NHWC)
    unsigned short* A_c1  = alloc_bf(8L * 192 * 256 * 64);
    unsigned short* A_c2  = alloc_bf(8L * 96 * 128 * 128);
    unsigned short* A_c3a = alloc_bf(8L * 48 * 64 * 256);
    unsigned short* A_c3b = alloc_bf(8L * 48 * 64 * 256);
    unsigned short* A_cat = alloc_bf(8L * 48 * 64 * 480);
    unsigned short* A_x31 = alloc_bf(8L * 48 * 64 * 256);
    unsigned short* A_x4  = alloc_bf(8L * 24 * 32 * 512);
    unsigned short* A_x41 = alloc_bf(8L * 24 * 32 * 512);
    unsigned short* A_x5  = alloc_bf(8L * 12 * 16 * 512);
    unsigned short* A_x51 = alloc_bf(8L * 12 * 16 * 512);
    unsigned short* A_x6  = alloc_bf(8L * 6 * 8 * 1024);
    unsigned short* A_x61 = alloc_bf(8L * 6 * 8 * 1024);

    auto cvt = [&](const float* src, unsigned short* dst, int Co, int Ci, int KHW, int Kc) {
        long total = (long)Co * KHW * Kc;
        cvt_weight<<<(int)((total + 255) / 256), 256, 0, stream>>>(src, dst, Co, Ci, KHW, Kc);
    };
    cvt(w2,  wb2,  128, 64, 25, 64);
    cvt(w3,  wb3,  256, 128, 25, 128);
    cvt(wr,  wbr,  32, 256, 1, 256);
    cvt(w31, wb31, 256, 473, 9, 480);   // pad K to 480 with zeros
    cvt(w4,  wb4,  512, 256, 9, 256);
    cvt(w41, wb41, 512, 512, 9, 512);
    cvt(w5,  wb5,  512, 512, 9, 512);
    cvt(w51, wb51, 512, 512, 9, 512);
    cvt(w6,  wb6,  1024, 512, 9, 512);
    cvt(w61, wb61, 1024, 1024, 9, 1024);

    auto conv = [&](const unsigned short* X, const unsigned short* Wt, const float* bias,
                    unsigned short* Y, float* Ynchw, int Hin, int Win, int Hout, int Wout,
                    int stride_, int pad_, int kh, int kw, int Cout, int Kc,
                    int out_cs, int out_choff) {
        int Npos = 8 * Hout * Wout;              // always a multiple of 128 here
        int tilesM = (Cout + 63) / 64;
        int tilesN = Npos / 128;
        conv_wmma<<<tilesM * tilesN, 256, 0, stream>>>(X, Wt, bias, Y, Ynchw,
                                                       Hin, Win, Hout, Wout, Npos,
                                                       stride_, pad_, kh, kw, Cout, Kc,
                                                       out_cs, out_choff);
    };

    // ---- image1 feature path ----
    {
        long n = 8L * 192 * 256 * 64;
        conv1_direct<<<(int)((n + 255) / 256), 256, 0, stream>>>(image1, w1, b1, A_c1);
    }
    conv(A_c1, wb2, b2, A_c2, out_c2a, 192, 256, 96, 128, 2, 2, 5, 5, 128, 64, 128, 0);
    conv(A_c2, wb3, b3, A_c3a, nullptr, 96, 128, 48, 64, 2, 2, 5, 5, 256, 128, 256, 0);
    // ---- image2 feature path (reuse c1/c2 buffers) ----
    {
        long n = 8L * 192 * 256 * 64;
        conv1_direct<<<(int)((n + 255) / 256), 256, 0, stream>>>(image2, w1, b1, A_c1);
    }
    conv(A_c1, wb2, b2, A_c2, nullptr, 192, 256, 96, 128, 2, 2, 5, 5, 128, 64, 128, 0);
    conv(A_c2, wb3, b3, A_c3b, nullptr, 96, 128, 48, 64, 2, 2, 5, 5, 256, 128, 256, 0);

    // ---- concat buffer: zero (for the 473->480 pad), redir, correlation ----
    {
        long ndw = 8L * 48 * 64 * 480 / 2;
        zero_u32<<<(int)((ndw + 255) / 256), 256, 0, stream>>>((unsigned int*)A_cat, ndw);
    }
    conv(A_c3a, wbr, br, A_cat, nullptr, 48, 64, 48, 64, 1, 0, 1, 1, 32, 256, 480, 0);
    correlation_k<<<8 * 48 * 64, 448, 0, stream>>>(A_c3a, A_c3b, A_cat);

    // ---- trunk ----
    conv(A_cat, wb31, b31, A_x31, out_x31, 48, 64, 48, 64, 1, 1, 3, 3, 256, 480, 256, 0);
    conv(A_x31, wb4,  b4,  A_x4,  nullptr, 48, 64, 24, 32, 2, 1, 3, 3, 512, 256, 512, 0);
    conv(A_x4,  wb41, b41, A_x41, out_x41, 24, 32, 24, 32, 1, 1, 3, 3, 512, 512, 512, 0);
    conv(A_x41, wb5,  b5,  A_x5,  nullptr, 24, 32, 12, 16, 2, 1, 3, 3, 512, 512, 512, 0);
    conv(A_x5,  wb51, b51, A_x51, out_x51, 12, 16, 12, 16, 1, 1, 3, 3, 512, 512, 512, 0);
    conv(A_x51, wb6,  b6,  A_x6,  nullptr, 12, 16, 6, 8, 2, 1, 3, 3, 1024, 512, 1024, 0);
    conv(A_x6,  wb61, b61, A_x61, out_x61, 6, 8, 6, 8, 1, 1, 3, 3, 1024, 1024, 1024, 0);
}
